// InpaintGenerator_9706626089973
// MI455X (gfx1250) — matrix operand order repaired and checked by reading, not compiled, not run on hardware
//
#include <hip/hip_runtime.h>
#include <hip/hip_bf16.h>
#include <math.h>

// ---------------------------------------------------------------------------
// CDNA5 (gfx1250) WMMA implementation of the InpaintGenerator forward pass.
// - v_wmma_f32_16x16x32_f16 (f32 accum) for all GEMM-shaped math
// - TDM tensor_load_to_lds (TENSORcnt) stages conv weight tiles
// - global_load_async_to_lds_b128 (ASYNCcnt) stages attention Q tiles
// ---------------------------------------------------------------------------

typedef _Float16 v16h __attribute__((ext_vector_type(16)));
typedef _Float16 v8h  __attribute__((ext_vector_type(8)));
typedef float    v8f  __attribute__((ext_vector_type(8)));
typedef int      v4i  __attribute__((vector_size(16)));
typedef unsigned int v4u __attribute__((vector_size(16)));
typedef int      v8i_ __attribute__((vector_size(32)));

union V16U { v16h v; v8h h[2]; _Float16 e[16]; };

__device__ __forceinline__ v8h ld8h(const _Float16* p) {
    v8h r;
    __builtin_memcpy(&r, p, sizeof(r));
    return r;
}

#if defined(__has_builtin)
#  if __has_builtin(__builtin_amdgcn_global_load_async_to_lds_b128)
#    define HAVE_ASYNC_LDS 1
#  else
#    define HAVE_ASYNC_LDS 0
#  endif
#  if __has_builtin(__builtin_amdgcn_tensor_load_to_lds)
#    define HAVE_TDM 1
#  else
#    define HAVE_TDM 0
#  endif
#  if __has_builtin(__builtin_amdgcn_s_wait_asynccnt)
#    define WAIT_ASYNC() __builtin_amdgcn_s_wait_asynccnt(0)
#  else
#    define WAIT_ASYNC() asm volatile("s_wait_asynccnt 0x0" ::: "memory")
#  endif
#  if __has_builtin(__builtin_amdgcn_s_wait_tensorcnt)
#    define WAIT_TENSOR() __builtin_amdgcn_s_wait_tensorcnt(0)
#  else
#    define WAIT_TENSOR() asm volatile("s_wait_tensorcnt 0x0" ::: "memory")
#  endif
#else
#  define HAVE_ASYNC_LDS 0
#  define HAVE_TDM 0
#  define WAIT_ASYNC()
#  define WAIT_TENSOR()
#endif

// async builtin signature: (v4i AS1* gsrc, v4i AS3* lds_dst, imm offset, imm cpol)
#define AS1_V4I(p) ((__attribute__((address_space(1))) v4i*)(void*)(p))
#define AS3_V4I(p) ((__attribute__((address_space(3))) v4i*)(p))

// ---------------------------------------------------------------------------
// f32 -> f16 conversion
// ---------------------------------------------------------------------------
__global__ void cvt_f32_f16_k(const float* __restrict__ in, _Float16* __restrict__ out, long n) {
    long i = (long)blockIdx.x * blockDim.x + threadIdx.x;
    if (i < n) out[i] = (_Float16)in[i];
}

// ---------------------------------------------------------------------------
// Implicit-GEMM convolution, 16x16 wave tiles, f16 WMMA, f32 accum.
//   A (16xK)  = weights (OIHW, contiguous along K = Cin*kh*kw), staged per
//               K-chunk into LDS by the Tensor Data Mover (16x32-half 2D tile,
//               row stride KK) -- falls back to async-load-to-LDS / ds_store
//   B (Kx16)  = im2col gather; k->(ci,ky,kx) decode pre-tabulated in LDS
//   act: 0 none, 1 leaky-relu(0.2), 2 tanh ; outF32: write f32 (else f16)
// blockDim = 128 (4 waves); wave w: columns nBase=(bx*4+w)*16, rows by*16
// ---------------------------------------------------------------------------
#define MAX_KK 2304   // Cin=256 * 3*3 (largest K in this net)

__global__ void conv_wmma_k(const _Float16* __restrict__ in, const _Float16* __restrict__ wt,
                            const float* __restrict__ bias, void* __restrict__ out,
                            int N, int Cin, int Hin, int Win,
                            int Cout, int Hout, int Wout,
                            int kh, int kw, int stride, int pad, int dil,
                            int act, int outF32) {
    __shared__ int   sOff[MAX_KK];     // ci * Hin * Win
    __shared__ short sDy[MAX_KK];      // ky*dil - pad
    __shared__ short sDx[MAX_KK];      // kx*dil - pad
    __shared__ _Float16 aTile[16 * 32];

    const int lane = threadIdx.x & 31;
    const int wave = threadIdx.x >> 5;
    const int col  = lane & 15;
    const int hi   = lane >> 4;           // 0: lanes 0-15, 1: lanes 16-31
    const int khkw = kh * kw;
    const int KK   = Cin * khkw;
    const int HinWin = Hin * Win;

    // ---- build k-decode tables once per block (div/mod hoisted out of K loop)
    for (int k = threadIdx.x; k < KK; k += blockDim.x) {
        int ci = k / khkw;
        int kr = k - ci * khkw;
        int ky = kr / kw;
        int kx = kr - ky * kw;
        sOff[k] = ci * HinWin;
        sDy[k]  = (short)(ky * dil - pad);
        sDx[k]  = (short)(kx * dil - pad);
    }
    __syncthreads();

    const long Npos = (long)N * Hout * Wout;
    const long nIdx = ((long)blockIdx.x * 4 + wave) * 16 + col;
    const bool nOK  = nIdx < Npos;

    int oys = 0, oxs = 0;
    size_t imgBase = 0;
    if (nOK) {
        int img = (int)(nIdx / ((long)Hout * Wout));
        int r   = (int)(nIdx % ((long)Hout * Wout));
        oys = (r / Wout) * stride;
        oxs = (r % Wout) * stride;
        imgBase = (size_t)img * Cin * HinWin;
    }

    const int  mBase  = blockIdx.y * 16;
    const bool fastM  = (mBase + 16 <= Cout);
    const bool fastK  = ((KK & 31) == 0);

    v8f acc;
#pragma unroll
    for (int i = 0; i < 8; i++) acc[i] = 0.f;

    if (fastM && fastK) {
        // ---------- fast path: TDM / async-stage A tile to LDS ----
        const int t = threadIdx.x;
#if HAVE_TDM
        const unsigned ldsByte =
            (unsigned)(unsigned long long)(__attribute__((address_space(3))) _Float16*)aTile;
#endif
        for (int kc = 0; kc < KK; kc += 32) {
#if HAVE_TDM
            if (wave == 0) {
                // D# for a 2D tile: 16 rows x 32 halves, row stride KK halves
                unsigned long long ga =
                    (unsigned long long)(const void*)(wt + (size_t)mBase * KK + kc);
                v4u g0 = { 1u,                                   // count=1 (valid D#)
                           ldsByte,                              // lds_addr
                           (unsigned)(ga & 0xFFFFFFFFu),         // global_addr[31:0]
                           (unsigned)(((ga >> 32) & 0x01FFFFFFull) | 0x80000000u) }; // [56:32] | type=2
                v8i_ g1 = { 0x10000,                             // data_size=1 (2 bytes)
                            (KK & 0xFFFF) << 16,                 // tensor_dim0[15:0]
                            ((KK >> 16) & 0xFFFF) | (16 << 16),  // tensor_dim0[31:16] | tensor_dim1=16
                            (32 << 16),                          // tile_dim0=32
                            16,                                  // tile_dim1=16
                            KK,                                  // tensor_dim0_stride[31:0]
                            0, 0 };
                v4i gz = { 0, 0, 0, 0 };
#  if __clang_major__ >= 23
                v8i_ gz8 = { 0, 0, 0, 0, 0, 0, 0, 0 };
                __builtin_amdgcn_tensor_load_to_lds(g0, g1, gz, gz, gz8, 0);
#  else
                __builtin_amdgcn_tensor_load_to_lds(g0, g1, gz, gz, 0);
#  endif
                WAIT_TENSOR();
            }
            __syncthreads();
#else
            // stage 16 rows x 32 halves (1 KB) with 64 lanes x b128
            if (t < 64) {
                const int row = t >> 2, q = t & 3;
                const _Float16* gsrc = wt + (size_t)(mBase + row) * KK + kc + q * 8;
                _Float16*       ldst = aTile + row * 32 + q * 8;
#  if HAVE_ASYNC_LDS
                __builtin_amdgcn_global_load_async_to_lds_b128(AS1_V4I(gsrc), AS3_V4I(ldst), 0, 0);
                WAIT_ASYNC();
#  else
                v8h tmp = ld8h(gsrc);
                __builtin_memcpy(ldst, &tmp, sizeof(tmp));
#  endif
            }
            __syncthreads();
#endif
            V16U A, B;
            const _Float16* arow = aTile + col * 32;
            A.h[0] = ld8h(arow + hi * 8);          // K = kc + hi*8 + 0..7
            A.h[1] = ld8h(arow + 16 + hi * 8);     // K = kc + 16 + hi*8 + 0..7

#pragma unroll
            for (int e = 0; e < 16; e++) {
                int k = kc + ((e < 8) ? (hi * 8 + e) : (16 + hi * 8 + (e - 8)));
                _Float16 val = (_Float16)0;
                if (nOK) {
                    int iy = oys + sDy[k];
                    int ix = oxs + sDx[k];
                    if ((unsigned)iy < (unsigned)Hin && (unsigned)ix < (unsigned)Win)
                        val = in[imgBase + sOff[k] + (size_t)iy * Win + ix];
                }
                B.e[e] = val;
            }

            acc = __builtin_amdgcn_wmma_f32_16x16x32_f16(false, A.v, false, B.v,
                                                         (short)0, acc, false, false);
            __syncthreads();   // aTile reused next chunk
        }
    } else {
        // ---------- tail path (Cout%16!=0 or KK%32!=0): guarded scalar A ----
        const int  mRow   = mBase + col;
        const bool mRowOK = mRow < Cout;
        const _Float16* wrow = wt + (size_t)mRow * KK;
        for (int kc = 0; kc < KK; kc += 32) {
            V16U A, B;
#pragma unroll
            for (int e = 0; e < 16; e++) {
                int k = kc + ((e < 8) ? (hi * 8 + e) : (16 + hi * 8 + (e - 8)));
                A.e[e] = (mRowOK && k < KK) ? wrow[k] : (_Float16)0;
            }
#pragma unroll
            for (int e = 0; e < 16; e++) {
                int k = kc + ((e < 8) ? (hi * 8 + e) : (16 + hi * 8 + (e - 8)));
                _Float16 val = (_Float16)0;
                if (nOK && k < KK) {
                    int iy = oys + sDy[k];
                    int ix = oxs + sDx[k];
                    if ((unsigned)iy < (unsigned)Hin && (unsigned)ix < (unsigned)Win)
                        val = in[imgBase + sOff[k] + (size_t)iy * Win + ix];
                }
                B.e[e] = val;
            }
            acc = __builtin_amdgcn_wmma_f32_16x16x32_f16(false, A.v, false, B.v,
                                                         (short)0, acc, false, false);
        }
    }

    if (!nOK) return;
    const int img = (int)(nIdx / ((long)Hout * Wout));
    const int rr  = (int)(nIdx % ((long)Hout * Wout));
    const int oy  = rr / Wout, ox = rr % Wout;
#pragma unroll
    for (int r = 0; r < 8; r++) {
        int m = mBase + r + hi * 8;            // D layout: VGPR r -> M = r (+8 hi half)
        if (m < Cout) {
            float v = acc[r] + bias[m];
            if (act == 1) v = (v >= 0.f) ? v : 0.2f * v;
            size_t o = (((size_t)img * Cout + m) * Hout + oy) * Wout + ox;
            if (outF32) ((float*)out)[o] = (act == 2) ? tanhf(v) : v;
            else        ((_Float16*)out)[o] = (_Float16)v;
        }
    }
}

// ---------------------------------------------------------------------------
// scores = Q(T,D) * K(T,D)^T * scale -> f32 (T,T).
// The Q row-tile (block-uniform) is async-staged into LDS and shared by all
// 4 waves; each wave computes TWO 16x16 N-tiles (2 wmma per K step).
// T multiple of 32, D of 32; all global loads are 128-bit.
// ---------------------------------------------------------------------------
__global__ void gemm_nt_scores_k(const _Float16* __restrict__ Q, const _Float16* __restrict__ Kt,
                                 float* __restrict__ S, int T, int D, float scale) {
    __shared__ _Float16 aT[16 * 32];
    const int t    = threadIdx.x;
    const int lane = t & 31;
    const int wave = t >> 5;
    const int col  = lane & 15;
    const int hi   = lane >> 4;
    const int mBase  = blockIdx.y * 16;
    const int nBase  = (blockIdx.x * 4 + wave) * 32;
    const bool active = nBase < T;

    const _Float16* bp0 = Kt + (size_t)(nBase + col) * D;
    const _Float16* bp1 = Kt + (size_t)(nBase + 16 + col) * D;

    v8f acc0, acc1;
#pragma unroll
    for (int i = 0; i < 8; i++) { acc0[i] = 0.f; acc1[i] = 0.f; }

    for (int kc = 0; kc < D; kc += 32) {
        if (t < 64) {   // stage Q rows mBase..mBase+15, K chunk, into LDS
            const int row = t >> 2, q = t & 3;
            const _Float16* gsrc = Q + (size_t)(mBase + row) * D + kc + q * 8;
            _Float16*       ldst = aT + row * 32 + q * 8;
#if HAVE_ASYNC_LDS
            __builtin_amdgcn_global_load_async_to_lds_b128(AS1_V4I(gsrc), AS3_V4I(ldst), 0, 0);
            WAIT_ASYNC();
#else
            v8h tmp = ld8h(gsrc);
            __builtin_memcpy(ldst, &tmp, sizeof(tmp));
#endif
        }
        __syncthreads();
        if (active) {
            V16U A, B0, B1;
            const _Float16* arow = aT + col * 32;
            A.h[0]  = ld8h(arow + hi * 8);
            A.h[1]  = ld8h(arow + 16 + hi * 8);
            B0.h[0] = ld8h(bp0 + kc + hi * 8);
            B0.h[1] = ld8h(bp0 + kc + 16 + hi * 8);
            B1.h[0] = ld8h(bp1 + kc + hi * 8);
            B1.h[1] = ld8h(bp1 + kc + 16 + hi * 8);
            __builtin_prefetch(bp0 + kc + 64, 0, 1);
            acc0 = __builtin_amdgcn_wmma_f32_16x16x32_f16(false, A.v, false, B0.v,
                                                          (short)0, acc0, false, false);
            acc1 = __builtin_amdgcn_wmma_f32_16x16x32_f16(false, A.v, false, B1.v,
                                                          (short)0, acc1, false, false);
        }
        __syncthreads();
    }
    if (!active) return;
#pragma unroll
    for (int r = 0; r < 8; r++) {
        size_t row = (size_t)(mBase + r + hi * 8) * T;
        S[row + nBase + col]      = acc0[r] * scale;
        S[row + nBase + 16 + col] = acc1[r] * scale;
    }
}

// ---------------------------------------------------------------------------
// Y(T,D) = attn(T,T) * V ; V supplied transposed as Vt(D,T). Two N-tiles/wave.
// T multiple of 32, D multiple of 128. f16 out.
// ---------------------------------------------------------------------------
__global__ void gemm_nn_attnv_k(const _Float16* __restrict__ A, const _Float16* __restrict__ Vt,
                                _Float16* __restrict__ Y, int T, int D) {
    const int lane = threadIdx.x & 31;
    const int wave = threadIdx.x >> 5;
    const int col  = lane & 15;
    const int hi   = lane >> 4;
    const int mBase = blockIdx.y * 16;                // token rows
    const int nBase = (blockIdx.x * 4 + wave) * 32;   // feature cols
    if (nBase >= D) return;

    const _Float16* ap  = A  + (size_t)(mBase + col) * T;
    const _Float16* bp0 = Vt + (size_t)(nBase + col) * T;
    const _Float16* bp1 = Vt + (size_t)(nBase + 16 + col) * T;

    v8f acc0, acc1;
#pragma unroll
    for (int i = 0; i < 8; i++) { acc0[i] = 0.f; acc1[i] = 0.f; }

    for (int kc = 0; kc < T; kc += 32) {
        V16U Aa, B0, B1;
        Aa.h[0] = ld8h(ap  + kc + hi * 8);
        Aa.h[1] = ld8h(ap  + kc + hi * 8 + 16);
        B0.h[0] = ld8h(bp0 + kc + hi * 8);
        B0.h[1] = ld8h(bp0 + kc + hi * 8 + 16);
        B1.h[0] = ld8h(bp1 + kc + hi * 8);
        B1.h[1] = ld8h(bp1 + kc + hi * 8 + 16);
        __builtin_prefetch(ap + kc + 64, 0, 1);
        acc0 = __builtin_amdgcn_wmma_f32_16x16x32_f16(false, Aa.v, false, B0.v,
                                                      (short)0, acc0, false, false);
        acc1 = __builtin_amdgcn_wmma_f32_16x16x32_f16(false, Aa.v, false, B1.v,
                                                      (short)0, acc1, false, false);
    }
#pragma unroll
    for (int r = 0; r < 8; r++) {
        size_t row = (size_t)(mBase + r + hi * 8) * D;
        Y[row + nBase + col]      = (_Float16)acc0[r];
        Y[row + nBase + 16 + col] = (_Float16)acc1[r];
    }
}

// ---------------------------------------------------------------------------
// Row softmax: f32 scores (T,T) -> f16 probabilities. One block per row.
// ---------------------------------------------------------------------------
__global__ void softmax_rows_k(const float* __restrict__ S, _Float16* __restrict__ P, int T) {
    __shared__ float red[256];
    const int row = blockIdx.x;
    const int tid = threadIdx.x;
    const float* srow = S + (size_t)row * T;

    float mx = -3.4e38f;
    for (int i = tid; i < T; i += 256) mx = fmaxf(mx, srow[i]);
    red[tid] = mx; __syncthreads();
    for (int s = 128; s > 0; s >>= 1) { if (tid < s) red[tid] = fmaxf(red[tid], red[tid + s]); __syncthreads(); }
    mx = red[0]; __syncthreads();

    float sum = 0.f;
    for (int i = tid; i < T; i += 256) sum += __expf(srow[i] - mx);
    red[tid] = sum; __syncthreads();
    for (int s = 128; s > 0; s >>= 1) { if (tid < s) red[tid] += red[tid + s]; __syncthreads(); }
    const float inv = 1.f / red[0];

    _Float16* prow = P + (size_t)row * T;
    for (int i = tid; i < T; i += 256)
        prow[i] = (_Float16)(__expf(srow[i] - mx) * inv);
}

// ---------------------------------------------------------------------------
// Patch tokenize: NCHW slice -> token matrix (T,D) or transposed (D,T).
// ---------------------------------------------------------------------------
__global__ void tokenize_k(const _Float16* __restrict__ src, _Float16* __restrict__ dst,
                           int transposed, int C, int H, int W, int grpBase,
                           int ph, int pw, int oh, int ow, int T, int D) {
    long idx = (long)blockIdx.x * blockDim.x + threadIdx.x;
    long tot = (long)T * D;
    if (idx >= tot) return;
    int tok = (int)(idx / D), d = (int)(idx % D);
    int ohow = oh * ow;
    int tt = tok / ohow; int r = tok % ohow; int oy = r / ow; int oxx = r % ow;
    int phpw = ph * pw;
    int c = d / phpw; int pr = d % phpw; int py = pr / pw; int px = pr % pw;
    _Float16 v = src[(((size_t)tt * C + grpBase + c) * H + (oy * ph + py)) * W + (oxx * pw + px)];
    if (transposed) dst[(size_t)d * T + tok] = v;
    else            dst[idx] = v;
}

__global__ void untokenize_k(const _Float16* __restrict__ Y, _Float16* __restrict__ dst,
                             int C, int H, int W, int grpBase,
                             int ph, int pw, int oh, int ow, int T, int D) {
    long idx = (long)blockIdx.x * blockDim.x + threadIdx.x;
    long tot = (long)T * D;
    if (idx >= tot) return;
    int tok = (int)(idx / D), d = (int)(idx % D);
    int ohow = oh * ow;
    int tt = tok / ohow; int r = tok % ohow; int oy = r / ow; int oxx = r % ow;
    int phpw = ph * pw;
    int c = d / phpw; int pr = d % phpw; int py = pr / pw; int px = pr % pw;
    dst[(((size_t)tt * C + grpBase + c) * H + (oy * ph + py)) * W + (oxx * pw + px)] = Y[idx];
}

// ---------------------------------------------------------------------------
// Bilinear x2 upsample (align_corners=True), f16.
// ---------------------------------------------------------------------------
__global__ void up2_k(const _Float16* __restrict__ in, _Float16* __restrict__ out,
                      int NC, int H, int W) {
    const int OH = 2 * H, OW = 2 * W;
    long idx = (long)blockIdx.x * blockDim.x + threadIdx.x;
    long tot = (long)NC * OH * OW;
    if (idx >= tot) return;
    int nc = (int)(idx / ((long)OH * OW));
    int r  = (int)(idx % ((long)OH * OW));
    int oy = r / OW, ox = r % OW;
    float ry = (H > 1) ? (float)oy * (float)(H - 1) / (float)(OH - 1) : 0.f;
    float rx = (W > 1) ? (float)ox * (float)(W - 1) / (float)(OW - 1) : 0.f;
    int y0 = (int)ry, x0 = (int)rx;
    int y1 = min(y0 + 1, H - 1), x1 = min(x0 + 1, W - 1);
    float fy = ry - (float)y0, fx = rx - (float)x0;
    const _Float16* p = in + (size_t)nc * H * W;
    float v00 = (float)p[(size_t)y0 * W + x0], v01 = (float)p[(size_t)y0 * W + x1];
    float v10 = (float)p[(size_t)y1 * W + x0], v11 = (float)p[(size_t)y1 * W + x1];
    float v = (v00 * (1.f - fx) + v01 * fx) * (1.f - fy) + (v10 * (1.f - fx) + v11 * fx) * fy;
    out[idx] = (_Float16)v;
}

__global__ void add_inplace_k(_Float16* __restrict__ a, const _Float16* __restrict__ b, long n) {
    long i = (long)blockIdx.x * blockDim.x + threadIdx.x;
    if (i < n) a[i] = (_Float16)((float)a[i] + (float)b[i]);
}

// ---------------------------------------------------------------------------
// Host orchestration
// ---------------------------------------------------------------------------
namespace {
struct P  { const float* w; const float* b; int wn; };
struct PH { const _Float16* w; const float* b; };
}

extern "C" void kernel_launch(void* const* d_in, const int* in_sizes, int n_in,
                              void* d_out, int out_size, void* d_ws, size_t ws_size,
                              hipStream_t stream) {
    (void)out_size; (void)ws_size; (void)n_in;

    // ---- parse parameter leaves (handles insertion-order and sorted-key flattening)
    P enc[4], dec[4];
    struct Blk { P q, k, v, out, ff1, ff2; } blk[6];
    int ii = 2;
    auto take = [&](P& p) {
        p.w = (const float*)d_in[ii]; p.wn = in_sizes[ii]; ii++;
        p.b = (const float*)d_in[ii]; ii++;
    };
    if (in_sizes[2] == 1728) {            // insertion order: enc, blocks(q,k,v,out,ff1,ff2), dec
        for (int e = 0; e < 4; e++) take(enc[e]);
        for (int b = 0; b < 6; b++) { take(blk[b].q); take(blk[b].k); take(blk[b].v);
                                      take(blk[b].out); take(blk[b].ff1); take(blk[b].ff2); }
        for (int d = 0; d < 4; d++) take(dec[d]);
    } else {                              // sorted keys: blocks(ff1,ff2,k,out,q,v), dec, enc
        for (int b = 0; b < 6; b++) { take(blk[b].ff1); take(blk[b].ff2); take(blk[b].k);
                                      take(blk[b].out); take(blk[b].q); take(blk[b].v); }
        for (int d = 0; d < 4; d++) take(dec[d]);
        for (int e = 0; e < 4; e++) take(enc[e]);
    }

    // ---- workspace bump allocator
    char* wsBase = (char*)d_ws;
    size_t wsOff = 0;
    auto wsAlloc = [&](size_t bytes) -> void* {
        wsOff = (wsOff + 255) & ~(size_t)255;
        void* p = wsBase + wsOff;
        wsOff += bytes;
        return p;
    };

    const size_t BUF_HALVES = 37748736ull;        // 8*256*96*192 (max f16 activation)
    const size_t QKV_HALVES = 9437184ull;         // 8*256*48*96
    const size_t TOK_HALVES = 2359296ull;         // T*D (identical for every group)
    const size_t ATT_ELEMS  = 4608ull * 4608ull;  // largest T*T

    _Float16* wpool = (_Float16*)wsAlloc(13ull * 1024 * 1024 * 2);
    _Float16* bufA  = (_Float16*)wsAlloc(BUF_HALVES * 2);
    _Float16* bufB  = (_Float16*)wsAlloc(BUF_HALVES * 2);
    _Float16* qbuf  = (_Float16*)wsAlloc(QKV_HALVES * 2);
    _Float16* kbuf  = (_Float16*)wsAlloc(QKV_HALVES * 2);
    _Float16* vbuf  = (_Float16*)wsAlloc(QKV_HALVES * 2);
    _Float16* Qtok  = (_Float16*)wsAlloc(TOK_HALVES * 2);
    _Float16* Ktok  = (_Float16*)wsAlloc(TOK_HALVES * 2);
    _Float16* VtT   = (_Float16*)wsAlloc(TOK_HALVES * 2);
    _Float16* Ytok  = (_Float16*)wsAlloc(TOK_HALVES * 2);
    _Float16* attnP = (_Float16*)wsAlloc(ATT_ELEMS * 2);
    float*    score = (float*)   wsAlloc(ATT_ELEMS * 4);

    // ---- convert all weights to f16 (fresh every call; deterministic)
    size_t wcur = 0;
    auto cvtW = [&](const P& p) -> const _Float16* {
        _Float16* dst = wpool + wcur;
        long n = p.wn;
        cvt_f32_f16_k<<<dim3((unsigned)((n + 255) / 256)), dim3(256), 0, stream>>>(p.w, dst, n);
        wcur += (size_t)((n + 15) & ~15);
        return dst;
    };
    PH encH[4], decH[4];
    struct BlkH { PH q, k, v, out, ff1, ff2; } blkH[6];
    for (int e = 0; e < 4; e++) encH[e] = { cvtW(enc[e]), enc[e].b };
    for (int d = 0; d < 4; d++) decH[d] = { cvtW(dec[d]), dec[d].b };
    for (int b = 0; b < 6; b++) {
        blkH[b].q   = { cvtW(blk[b].q),   blk[b].q.b   };
        blkH[b].k   = { cvtW(blk[b].k),   blk[b].k.b   };
        blkH[b].v   = { cvtW(blk[b].v),   blk[b].v.b   };
        blkH[b].out = { cvtW(blk[b].out), blk[b].out.b };
        blkH[b].ff1 = { cvtW(blk[b].ff1), blk[b].ff1.b };
        blkH[b].ff2 = { cvtW(blk[b].ff2), blk[b].ff2.b };
    }

    // ---- conv launch helper
    auto conv = [&](const _Float16* in, const PH& p, void* out,
                    int N, int Cin, int Hin, int Win, int Cout,
                    int kh, int kw, int stride, int pad, int dil,
                    int act, int outF32) {
        int Hout = (Hin + 2 * pad - dil * (kh - 1) - 1) / stride + 1;
        int Wout = (Win + 2 * pad - dil * (kw - 1) - 1) / stride + 1;
        long Npos = (long)N * Hout * Wout;
        dim3 grid((unsigned)((Npos + 63) / 64), (unsigned)((Cout + 15) / 16));
        conv_wmma_k<<<grid, dim3(128), 0, stream>>>(in, p.w, p.b, out,
            N, Cin, Hin, Win, Cout, Hout, Wout, kh, kw, stride, pad, dil, act, outF32);
    };

    const int N = 8;

    // ---- input frames f32 -> f16 (N,3,192,384)
    {
        long n = (long)N * 3 * 192 * 384;
        cvt_f32_f16_k<<<dim3((unsigned)((n + 255) / 256)), dim3(256), 0, stream>>>(
            (const float*)d_in[0], bufA, n);
    }

    // ---- encoder: strides (2,1,2,1), all leaky-relu
    conv(bufA, encH[0], bufB, N,   3, 192, 384,  64, 3, 3, 2, 1, 1, 1, 0); // -> 96x192
    conv(bufB, encH[1], bufA, N,  64,  96, 192,  64, 3, 3, 1, 1, 1, 1, 0);
    conv(bufA, encH[2], bufB, N,  64,  96, 192, 128, 3, 3, 2, 1, 1, 1, 0); // -> 48x96
    conv(bufB, encH[3], bufA, N, 128,  48,  96, 256, 3, 3, 1, 1, 1, 1, 0);

    // ---- transformer blocks; x lives in bufA (N,256,48,96)
    const int HH = 48, WW = 96, C = 256, dk = 64;
    const int grpPW[4] = {48, 16, 8, 4};
    const int grpPH[4] = {24,  8, 4, 2};
    const long xN = (long)N * C * HH * WW;

    for (int b = 0; b < 6; b++) {
        // q/k/v 1x1 convs (no activation)
        conv(bufA, blkH[b].q, qbuf, N, C, HH, WW, C, 1, 1, 1, 0, 1, 0, 0);
        conv(bufA, blkH[b].k, kbuf, N, C, HH, WW, C, 1, 1, 1, 0, 1, 0, 0);
        conv(bufA, blkH[b].v, vbuf, N, C, HH, WW, C, 1, 1, 1, 0, 1, 0, 0);

        // per-group patch attention; concat result into bufB
        for (int g = 0; g < 4; g++) {
            const int pw = grpPW[g], ph = grpPH[g];
            const int ow = WW / pw, oh = HH / ph;
            const int T = N * oh * ow;
            const int D = dk * ph * pw;
            const float scale = 1.f / sqrtf((float)D);
            const long tot = (long)T * D;
            const unsigned tg = (unsigned)((tot + 255) / 256);

            tokenize_k<<<dim3(tg), dim3(256), 0, stream>>>(qbuf, Qtok, 0, C, HH, WW, g * dk, ph, pw, oh, ow, T, D);
            tokenize_k<<<dim3(tg), dim3(256), 0, stream>>>(kbuf, Ktok, 0, C, HH, WW, g * dk, ph, pw, oh, ow, T, D);
            tokenize_k<<<dim3(tg), dim3(256), 0, stream>>>(vbuf, VtT,  1, C, HH, WW, g * dk, ph, pw, oh, ow, T, D);

            gemm_nt_scores_k<<<dim3((unsigned)((T + 127) / 128), (unsigned)(T / 16)),
                               dim3(128), 0, stream>>>(Qtok, Ktok, score, T, D, scale);
            softmax_rows_k<<<dim3((unsigned)T), dim3(256), 0, stream>>>(score, attnP, T);
            gemm_nn_attnv_k<<<dim3((unsigned)((D + 127) / 128), (unsigned)(T / 16)),
                              dim3(128), 0, stream>>>(attnP, VtT, Ytok, T, D);

            untokenize_k<<<dim3(tg), dim3(256), 0, stream>>>(Ytok, bufB, C, HH, WW, g * dk, ph, pw, oh, ow, T, D);
        }

        // out conv 3x3 + lrelu, then residual add
        conv(bufB, blkH[b].out, qbuf, N, C, HH, WW, C, 3, 3, 1, 1, 1, 1, 0);
        add_inplace_k<<<dim3((unsigned)((xN + 255) / 256)), dim3(256), 0, stream>>>(bufA, qbuf, xN);

        // feed-forward: ff1 (dil=2,pad=2) + ff2, both lrelu, then residual add
        conv(bufA, blkH[b].ff1, bufB, N, C, HH, WW, C, 3, 3, 1, 2, 2, 1, 0);
        conv(bufB, blkH[b].ff2, qbuf, N, C, HH, WW, C, 3, 3, 1, 1, 1, 1, 0);
        add_inplace_k<<<dim3((unsigned)((xN + 255) / 256)), dim3(256), 0, stream>>>(bufA, qbuf, xN);
    }

    // ---- decoder
    {   // up2: (N,256,48,96) -> (N,256,96,192)
        long tot = (long)N * 256 * 96 * 192;
        up2_k<<<dim3((unsigned)((tot + 255) / 256)), dim3(256), 0, stream>>>(bufA, bufB, N * 256, 48, 96);
    }
    conv(bufB, decH[0], bufA, N, 256, 96, 192, 128, 3, 3, 1, 1, 1, 1, 0);
    conv(bufA, decH[1], bufB, N, 128, 96, 192,  64, 3, 3, 1, 1, 1, 1, 0);
    {   // up2: (N,64,96,192) -> (N,64,192,384)
        long tot = (long)N * 64 * 192 * 384;
        up2_k<<<dim3((unsigned)((tot + 255) / 256)), dim3(256), 0, stream>>>(bufB, bufA, N * 64, 96, 192);
    }
    conv(bufA, decH[2], bufB, N, 64, 192, 384, 64, 3, 3, 1, 1, 1, 1, 0);
    // final conv: tanh, f32 output straight into d_out (N,3,192,384)
    conv(bufB, decH[3], d_out, N, 64, 192, 384, 3, 3, 3, 1, 1, 1, 2, 1);
}
